// AttentionPruneKV_76613626626349
// MI455X (gfx1250) — matrix operand order repaired and checked by reading, not compiled, not run on hardware
//
#include <hip/hip_runtime.h>
#include <hip/hip_bf16.h>

// ---------------- problem constants ----------------
#define HEIGHT   32
#define WIDTH    32
#define NTOK     1024          // HEIGHT*WIDTH
#define HEADS    8
#define DHEAD    64
#define DIM      512
#define INNER    512
#define BATCH    8
#define QKV3     1536
#define ATT_SCALE 0.125f       // DHEAD^-0.5
#define NEG_TH   0.9f

typedef __bf16 bf16_t;
typedef __attribute__((ext_vector_type(16))) __bf16 v16bf;
typedef __attribute__((ext_vector_type(8)))  __bf16 v8bf;
typedef __attribute__((ext_vector_type(8)))  float  v8f;
typedef int v4i __attribute__((vector_size(16)));   // matches builtin param type

#if __has_builtin(__builtin_amdgcn_global_load_async_to_lds_b128) && \
    __has_builtin(__builtin_amdgcn_s_wait_asynccnt)
#define USE_ASYNC_LDS 1
#else
#define USE_ASYNC_LDS 0
#endif

__device__ __forceinline__ v8f wmma_bf16(v16bf a, v16bf b, v8f c) {
  // D = A(16x32 bf16) x B(32x16 bf16) + C(16x16 f32)
  return __builtin_amdgcn_wmma_f32_16x16x32_bf16(
      false, a, false, b, (short)0, c, false, false);
}

__device__ __forceinline__ float xshfl(float v, int mask) {
  return __shfl_xor(v, mask, 32);
}

// A-fragment (16x32, row = lane&15): two contiguous 16B chunks
__device__ __forceinline__ v16bf load_a_frag(const bf16_t* __restrict__ row, int half) {
  const v8bf lo = *(const v8bf*)(row + 8 * half);        // K = 8*half+e
  const v8bf hi = *(const v8bf*)(row + 16 + 8 * half);   // K = 16+8*half+e
  v16bf r;
#pragma unroll
  for (int e = 0; e < 8; ++e) { r[e] = lo[e]; r[e + 8] = hi[e]; }
  return r;
}

// B-fragment (32x16, col = lane&15): two contiguous 16B chunks (K = 8*half+e, 16+8*half+e)
__device__ __forceinline__ v16bf load_b_frag(const bf16_t* __restrict__ colbase, int half) {
  const v8bf lo = *(const v8bf*)(colbase + 8 * half);
  const v8bf hi = *(const v8bf*)(colbase + 16 + 8 * half);
  v16bf r;
#pragma unroll
  for (int e = 0; e < 8; ++e) { r[e] = lo[e]; r[e + 8] = hi[e]; }
  return r;
}

// ============================================================
// Prep kernels: bf16 conversion + weight transposes
// ============================================================
__global__ void cvt_x_kernel(const float* __restrict__ x, bf16_t* __restrict__ xb, int n) {
  const int i = blockIdx.x * 256 + threadIdx.x;
  if (i < n) xb[i] = (bf16_t)x[i];
}

// w[rows][cols] f32 -> wT[cols][rows] bf16
__global__ void transpose_w_kernel(const float* __restrict__ w, bf16_t* __restrict__ wT,
                                   int rows, int cols) {
  const int i = blockIdx.x * 256 + threadIdx.x;
  if (i >= rows * cols) return;
  const int r = i / cols, c = i % cols;
  wT[(size_t)c * rows + r] = (bf16_t)w[i];
}

// ============================================================
// Kernel A: qkv = x @ w_qkv -> q/k row-major [b,h,n,d], V transposed [b,h,d,n]
// ============================================================
__global__ __launch_bounds__(32)
void qkv_proj_kernel(const bf16_t* __restrict__ xb, const bf16_t* __restrict__ wqkvT,
                     bf16_t* __restrict__ qb, bf16_t* __restrict__ kb,
                     bf16_t* __restrict__ vT) {
  const int lane  = threadIdx.x & 31;
  const int l16   = lane & 15;
  const int half  = lane >> 4;
  const int ntile = blockIdx.x;   // 0..63 token tile
  const int ctile = blockIdx.y;   // 0..95 output-col tile
  const int b     = blockIdx.z;

  const int tok = ntile * 16 + l16;
  const int col = ctile * 16 + l16;
  const bf16_t* xrow = xb + ((size_t)b * NTOK + tok) * DIM;
  const bf16_t* wcol = wqkvT + (size_t)col * DIM;

  v8f acc = {};
  for (int k0 = 0; k0 < DIM; k0 += 32) {
    const v16bf a  = load_a_frag(xrow + k0, half);
    const v16bf bm = load_b_frag(wcol + k0, half);
    acc = wmma_bf16(a, bm, acc);
  }

  const int sec   = col / INNER;          // 0=q 1=k 2=v
  const int inner = col % INNER;
  const int hh    = inner / DHEAD;
  const int dh    = inner % DHEAD;
  if (sec == 2) {
    // transposed V: rows r map to consecutive tokens -> one packed 16B store
    v8bf pk;
#pragma unroll
    for (int r = 0; r < 8; ++r) pk[r] = (bf16_t)acc[r];
    *(v8bf*)(vT + (((size_t)b * HEADS + hh) * DHEAD + dh) * NTOK +
             ntile * 16 + 8 * half) = pk;
  } else {
    bf16_t* dst = (sec == 0) ? qb : kb;
#pragma unroll
    for (int r = 0; r < 8; ++r) {
      const int m = ntile * 16 + r + 8 * half;
      dst[(((size_t)b * HEADS + hh) * NTOK + m) * DHEAD + dh] = (bf16_t)acc[r];
    }
  }
}

// ============================================================
// Kernel B: t[b,h,n] = sigmoid(q . w_thresh) * sigmoid(gate)
// ============================================================
__global__ void thresh_kernel(const bf16_t* __restrict__ qb,
                              const float* __restrict__ w_thresh,
                              const float* __restrict__ gate,
                              float* __restrict__ tq) {
  const int idx = blockIdx.x * blockDim.x + threadIdx.x;
  if (idx >= BATCH * HEADS * NTOK) return;
  const bf16_t* qrow = qb + (size_t)idx * DHEAD;
  float s = 0.f;
#pragma unroll 4
  for (int d = 0; d < DHEAD; ++d) s += (float)qrow[d] * w_thresh[d];
  const float t = 1.f / (1.f + __expf(-s));
  const float g = 1.f / (1.f + __expf(-gate[0]));
  tq[idx] = t * g;
}

// ============================================================
// Kernel C: fused attention, one wave per (b,h,16-query) strip.
// K tiles streamed into LDS via async-to-LDS double buffering
// (falls back to direct b128 global loads if the builtin is absent).
// ============================================================
__global__ __launch_bounds__(32)
void attn_kernel(const bf16_t* __restrict__ qb, const bf16_t* __restrict__ kb,
                 const bf16_t* __restrict__ vT,
                 const float* __restrict__ rpb_table,
                 const float* __restrict__ headsita,
                 const float* __restrict__ prob,
                 const float* __restrict__ tq,
                 bf16_t* __restrict__ hidden,
                 float* __restrict__ attn0_out) {
  __shared__ bf16_t snum[16 * NTOK];            // 32 KB pruned-numerator strip
#if USE_ASYNC_LDS
  __shared__ bf16_t kstage[2][16 * DHEAD];      // 2 x 2KB K-tile double buffer
#endif

  const int lane  = threadIdx.x & 31;
  const int l16   = lane & 15;
  const int half  = lane >> 4;
  const int qt    = blockIdx.x;
  const int h     = blockIdx.y;
  const int b     = blockIdx.z;
  const int bh    = b * HEADS + h;
  const int qbase = qt * 16;

  const float sita   = headsita[h];
  const float factor = 1.f / (2.f * sita * sita + 1e-6f);

  const bf16_t* ktilebase = kb + (size_t)bh * NTOK * DHEAD;

#if USE_ASYNC_LDS
  // stage one 16x64 bf16 K tile (2048B): 32 lanes x 16B x 4 instructions
#define STAGE_K(j, buf)                                                        \
  {                                                                            \
    const char* srcb = (const char*)(ktilebase + (size_t)(j) * 16 * DHEAD);    \
    char*       dstb = (char*)&kstage[(buf)][0];                               \
    _Pragma("unroll")                                                          \
    for (int i_ = 0; i_ < 4; ++i_) {                                           \
      const int off_ = (i_ * 32 + lane) * 16;                                  \
      __builtin_amdgcn_global_load_async_to_lds_b128(                          \
          (__attribute__((address_space(1))) v4i*)(srcb + off_),               \
          (__attribute__((address_space(3))) v4i*)(dstb + off_), 0, 0);        \
    }                                                                          \
  }
#endif

  // ---- preload Q A-fragments ----
  const bf16_t* qrow = qb + ((size_t)bh * NTOK + qbase + l16) * DHEAD;
  const v16bf qa0 = load_a_frag(qrow, half);
  const v16bf qa1 = load_a_frag(qrow + 32, half);

  int qtok[8], qr[8], qc[8];
#pragma unroll
  for (int r = 0; r < 8; ++r) {
    qtok[r] = qbase + r + 8 * half;
    qr[r]   = qtok[r] >> 5;
    qc[r]   = qtok[r] & 31;
  }

  // ---------------- pass 1: online stats ----------------
  float m1[8], z1[8], mn1[8], m0[8], z0[8];
#pragma unroll
  for (int r = 0; r < 8; ++r) { m1[r] = -1e30f; z1[r] = 0.f; mn1[r] = 1e30f;
                                m0[r] = -1e30f; z0[r] = 0.f; }

#if USE_ASYNC_LDS
  STAGE_K(0, 0);
#endif
  for (int j = 0; j < 64; ++j) {
#if USE_ASYNC_LDS
    const int buf = j & 1;
    if (j < 63) { STAGE_K(j + 1, buf ^ 1); __builtin_amdgcn_s_wait_asynccnt(4); }
    else        { __builtin_amdgcn_s_wait_asynccnt(0); }
    const bf16_t* krow = &kstage[buf][l16 * DHEAD];
#else
    const bf16_t* krow = ktilebase + ((size_t)j * 16 + l16) * DHEAD;
#endif
    const v16bf kb0 = load_b_frag(krow, half);
    const v16bf kb1 = load_b_frag(krow + 32, half);
    v8f s = {};
    s = wmma_bf16(qa0, kb0, s);
    s = wmma_bf16(qa1, kb1, s);

    const int kt = j * 16 + l16;
    const int rk = kt >> 5, ck = kt & 31;
#pragma unroll
    for (int r = 0; r < 8; ++r) {
      const float d0 = s[r] * ATT_SCALE;
      const int dr = qr[r] - rk, dc = qc[r] - ck;
      const float relb = rpb_table[(size_t)((dr + 31) * 63 + (dc + 31)) * HEADS + h];
      const float fr = dr * (1.f / 32.f), fc = dc * (1.f / 32.f);
      const float dd = d0 + relb + 0.01f * __expf(-factor * (fr * fr + fc * fc));
      float mo = m1[r], mnew = fmaxf(mo, dd);
      z1[r] = z1[r] * __expf(mo - mnew) + __expf(dd - mnew);
      m1[r] = mnew;
      mn1[r] = fminf(mn1[r], dd);
      mo = m0[r]; mnew = fmaxf(mo, d0);
      z0[r] = z0[r] * __expf(mo - mnew) + __expf(d0 - mnew);
      m0[r] = mnew;
    }
  }
  // butterfly merge across the 16 lanes of each half-group
#pragma unroll
  for (int mask = 1; mask < 16; mask <<= 1) {
#pragma unroll
    for (int r = 0; r < 8; ++r) {
      float om = xshfl(m1[r], mask), oz = xshfl(z1[r], mask);
      float mnew = fmaxf(m1[r], om);
      z1[r] = z1[r] * __expf(m1[r] - mnew) + oz * __expf(om - mnew);
      m1[r] = mnew;
      mn1[r] = fminf(mn1[r], xshfl(mn1[r], mask));
      om = xshfl(m0[r], mask); oz = xshfl(z0[r], mask);
      mnew = fmaxf(m0[r], om);
      z0[r] = z0[r] * __expf(m0[r] - mnew) + oz * __expf(om - mnew);
      m0[r] = mnew;
    }
  }

  float thr[8], invz1[8], invz0[8];
#pragma unroll
  for (int r = 0; r < 8; ++r) {
    const float Z    = z1[r];
    const float amax = 1.f / Z;
    const float amin = __expf(mn1[r] - m1[r]) / Z;
    const float t    = tq[(size_t)bh * NTOK + qtok[r]];
    thr[r]   = amin + t * (amax - amin);
    invz1[r] = amax;
    invz0[r] = 1.f / z0[r];
  }

  // ---------------- pass 2: attn0 out + pruned numerator ----------------
  float deno[8];
#pragma unroll
  for (int r = 0; r < 8; ++r) deno[r] = 0.f;
  float* a0out = attn0_out + ((size_t)bh * NTOK + qbase) * NTOK;

#if USE_ASYNC_LDS
  STAGE_K(0, 0);
#endif
  for (int j = 0; j < 64; ++j) {
#if USE_ASYNC_LDS
    const int buf = j & 1;
    if (j < 63) { STAGE_K(j + 1, buf ^ 1); __builtin_amdgcn_s_wait_asynccnt(4); }
    else        { __builtin_amdgcn_s_wait_asynccnt(0); }
    const bf16_t* krow = &kstage[buf][l16 * DHEAD];
#else
    const bf16_t* krow = ktilebase + ((size_t)j * 16 + l16) * DHEAD;
#endif
    const v16bf kb0 = load_b_frag(krow, half);
    const v16bf kb1 = load_b_frag(krow + 32, half);
    v8f s = {};
    s = wmma_bf16(qa0, kb0, s);
    s = wmma_bf16(qa1, kb1, s);

    const int kt = j * 16 + l16;
    const int rk = kt >> 5, ck = kt & 31;
#pragma unroll
    for (int r = 0; r < 8; ++r) {
      const float d0 = s[r] * ATT_SCALE;
      const int dr = qr[r] - rk, dc = qc[r] - ck;
      const float relb = rpb_table[(size_t)((dr + 31) * 63 + (dc + 31)) * HEADS + h];
      const float fr = dr * (1.f / 32.f), fc = dc * (1.f / 32.f);
      const float dd = d0 + relb + 0.01f * __expf(-factor * (fr * fr + fc * fc));

      const float attn  = __expf(dd - m1[r]) * invz1[r];
      const float attn0 = __expf(d0 - m0[r]) * invz0[r];
      const int   M     = r + 8 * half;
      a0out[(size_t)M * NTOK + kt] = attn0;           // required 2nd output

      const float num = (attn > thr[r]) ? attn : 0.f; // attn * record
      deno[r] += num;
      snum[M * NTOK + kt] = (bf16_t)num;
    }
  }
#pragma unroll
  for (int mask = 1; mask < 16; mask <<= 1)
#pragma unroll
    for (int r = 0; r < 8; ++r) deno[r] += xshfl(deno[r], mask);

  float rscale[8];
#pragma unroll
  for (int r = 0; r < 8; ++r) {
    const float pv = prob[(size_t)b * NTOK + qtok[r]];
    rscale[r] = (pv >= NEG_TH) ? 0.f : 1.f / (deno[r] + 1e-6f);
  }

  // ---------------- pass 3: numerator @ V (V transposed: contiguous K) ----
  v8f acc[4];
#pragma unroll
  for (int t = 0; t < 4; ++t) acc[t] = (v8f){};

  const bf16_t* vbase = vT + (size_t)bh * DHEAD * NTOK;
  for (int k0 = 0; k0 < NTOK; k0 += 32) {
    if (k0 + 32 < NTOK)
      __builtin_prefetch(vbase + (size_t)l16 * NTOK + k0 + 32, 0, 1);
    const v16bf a = load_a_frag(snum + l16 * NTOK + k0, half);
#pragma unroll
    for (int nt = 0; nt < 4; ++nt) {
      const int dcol = nt * 16 + l16;
      const v16bf bv = load_b_frag(vbase + (size_t)dcol * NTOK + k0, half);
      acc[nt] = wmma_bf16(a, bv, acc[nt]);
    }
  }

#pragma unroll
  for (int nt = 0; nt < 4; ++nt)
#pragma unroll
    for (int r = 0; r < 8; ++r) {
      const int tok  = qbase + r + 8 * half;
      const int dcol = nt * 16 + l16;
      hidden[((size_t)b * NTOK + tok) * INNER + h * DHEAD + dcol] =
          (bf16_t)(acc[nt][r] * rscale[r]);
    }
}

// ============================================================
// Kernel D: out = hidden @ w_out + b_out   (f32 result)
// ============================================================
__global__ __launch_bounds__(32)
void out_proj_kernel(const bf16_t* __restrict__ hidden,
                     const bf16_t* __restrict__ woutT,
                     const float* __restrict__ b_out,
                     float* __restrict__ out) {
  const int lane = threadIdx.x & 31;
  const int l16  = lane & 15;
  const int half = lane >> 4;
  const int rt   = blockIdx.x;    // 0..511
  const int ct   = blockIdx.y;    // 0..31
  const int row  = rt * 16 + l16;
  const int col  = ct * 16 + l16;
  const bf16_t* hrow = hidden + (size_t)row * INNER;
  const bf16_t* wcol = woutT + (size_t)col * INNER;

  v8f acc = {};
  for (int k0 = 0; k0 < INNER; k0 += 32) {
    const v16bf a  = load_a_frag(hrow + k0, half);
    const v16bf bv = load_b_frag(wcol + k0, half);
    acc = wmma_bf16(a, bv, acc);
  }
  const float bias = b_out[col];
#pragma unroll
  for (int r = 0; r < 8; ++r) {
    const int m = rt * 16 + r + 8 * half;
    out[(size_t)m * DIM + col] = acc[r] + bias;
  }
}

// ============================================================
extern "C" void kernel_launch(void* const* d_in, const int* in_sizes, int n_in,
                              void* d_out, int out_size, void* d_ws, size_t ws_size,
                              hipStream_t stream) {
  const float* x        = (const float*)d_in[0];
  const float* prob     = (const float*)d_in[1];
  const float* w_qkv    = (const float*)d_in[2];
  const float* rpb      = (const float*)d_in[3];
  const float* headsita = (const float*)d_in[4];
  const float* w_thresh = (const float*)d_in[5];
  const float* gate     = (const float*)d_in[6];
  const float* w_out    = (const float*)d_in[7];
  const float* b_out    = (const float*)d_in[8];

  const size_t qkv_elems = (size_t)BATCH * HEADS * NTOK * DHEAD;   // 4 Mi
  bf16_t* qb     = (bf16_t*)d_ws;
  bf16_t* kb     = qb + qkv_elems;
  bf16_t* vT     = kb + qkv_elems;                  // [b,h,d,n]
  float*  tq     = (float*)(vT + qkv_elems);
  bf16_t* hidden = (bf16_t*)(tq + (size_t)BATCH * HEADS * NTOK);
  bf16_t* xb     = hidden + (size_t)BATCH * NTOK * INNER;
  bf16_t* wqkvT  = xb + (size_t)BATCH * NTOK * DIM;
  bf16_t* woutT  = wqkvT + (size_t)QKV3 * DIM;

  float* out   = (float*)d_out;
  float* attn0 = out + (size_t)BATCH * NTOK * DIM;

  const int nx = BATCH * NTOK * DIM;
  cvt_x_kernel<<<dim3((nx + 255) / 256), 256, 0, stream>>>(x, xb, nx);
  transpose_w_kernel<<<dim3((DIM * QKV3 + 255) / 256), 256, 0, stream>>>(
      w_qkv, wqkvT, DIM, QKV3);
  transpose_w_kernel<<<dim3((INNER * DIM + 255) / 256), 256, 0, stream>>>(
      w_out, woutT, INNER, DIM);

  qkv_proj_kernel<<<dim3(NTOK / 16, QKV3 / 16, BATCH), 32, 0, stream>>>(
      xb, wqkvT, qb, kb, vT);
  thresh_kernel<<<dim3(BATCH * HEADS * NTOK / 256), 256, 0, stream>>>(
      qb, w_thresh, gate, tq);
  attn_kernel<<<dim3(NTOK / 16, HEADS, BATCH), 32, 0, stream>>>(
      qb, kb, vT, rpb, headsita, prob, tq, hidden, attn0);
  out_proj_kernel<<<dim3(BATCH * NTOK / 16, DIM / 16), 32, 0, stream>>>(
      hidden, woutT, b_out, out);
}
